// GNN_28724741276435
// MI455X (gfx1250) — compile-verified, matrix operand order checked by reading, not run on hardware
//
#include <hip/hip_runtime.h>
#include <math.h>

#define BATCH 2
#define HGT   64
#define WID   64
#define HWSZ  (HGT*WID)      // 4096
#define BHW   (BATCH*HWSZ)   // 8192
#define HID   10
#define IND   256

typedef __attribute__((ext_vector_type(16))) _Float16 v16h;
typedef __attribute__((ext_vector_type(8)))  float    v8f;

static inline int ceil_div(int a, int b) { return (a + b - 1) / b; }

__device__ __forceinline__ void wait_async0() {
#if __has_builtin(__builtin_amdgcn_s_wait_asynccnt)
  __builtin_amdgcn_s_wait_asynccnt(0);
#else
  asm volatile("s_wait_asynccnt 0x0" ::: "memory");
#endif
}

// Issue one per-lane async global->LDS 16B copy (ASYNCcnt-tracked).
__device__ __forceinline__ void async_g2l_b128(unsigned lds_off, const float* gaddr) {
  asm volatile("global_load_async_to_lds_b128 %0, %1, off"
               :: "v"(lds_off), "v"(gaddr) : "memory");
}

// ---------------------------------------------------------------------------
// Implicit-GEMM convolution, f32 -> f16 WMMA -> f32.
//   out[b,n,h,w] = epilogue( sum_{c,ky,kx} x[b,c,h+ky-pad,w+kx-pad] * wgt[n, c*KK+kidx] )
// act: 0 -> v + bias[n]; 1 -> relu(g[n]*v + bias[n]); 2 -> sigmoid(v + bias[n])
// prescale: optional per-pixel scalar folded into A rows at fragment build.
// Tiling: 128(M) x 32(N) per block, 8 waves; wave wv owns M rows
// [wv*16, wv*16+16) and BOTH 16-col N slices (two accumulators, A-fragment
// reused across two WMMAs). K chunk = 64 -> 4 WMMAs per chunk.
// A/B staged K-major in LDS as f32; for ks==1 with a fully in-range chunk the
// A tile is staged with global_load_async_to_lds_b128 (16B per lane-op).
// ---------------------------------------------------------------------------
__global__ __launch_bounds__(256)
void conv_wmma_kernel(const float* __restrict__ x, const float* __restrict__ wgt,
                      const float* __restrict__ gscale, const float* __restrict__ bias,
                      const float* __restrict__ prescale, float* __restrict__ out,
                      int Cin, int N, int ks, int act)
{
  const int KK   = ks * ks;
  const int pad  = ks >> 1;
  const int Ktot = Cin * KK;
  const int mtile = blockIdx.x * 128;
  const int ntile = blockIdx.y * 32;

  __shared__ __align__(16) float As[64 * 132];  // [kk][mi], stride 132
  __shared__ __align__(16) float Bs[64 * 36];   // [kk][ni], stride 36

  const int tid  = threadIdx.x;
  const int lane = tid & 31;
  const int wv   = tid >> 5;
  const int mrow = wv * 16 + (lane & 15);
  const int nc0  = lane & 15;
  const int nc1  = 16 + nc0;
  const int kofs = (lane >= 16) ? 8 : 0;

  const float psc = prescale ? prescale[mtile + mrow] : 1.0f;

  v8f acc0 = {}, acc1 = {};

  for (int kb = 0; kb < Ktot; kb += 64) {
    const bool useAsync = (ks == 1) && (kb + 64 <= Ktot);
    if (useAsync) {
      // 64 K-rows x 128 pixels, 16B per op: 2048 ops / 256 threads.
      #pragma unroll
      for (int it = 0; it < 8; ++it) {
        int i   = tid + it * 256;
        int kk  = i >> 5;
        int mi4 = (i & 31) << 2;
        int m = mtile + mi4;
        int b = m / HWSZ, hw = m - b * HWSZ;
        int c = kb + kk;
        const float* gp = x + ((size_t)b * Cin + c) * HWSZ + hw;
        unsigned lp = (unsigned)(unsigned long long)(&As[kk * 132 + mi4]);
        async_g2l_b128(lp, gp);
      }
    } else {
      for (int it = 0; it < 32; ++it) {
        int i  = tid + it * 256;
        int kk = i >> 7, mi = i & 127;
        int m = mtile + mi;
        int k = kb + kk;
        float v = 0.0f;
        if (k < Ktot) {
          int b  = m / HWSZ, hw = m - b * HWSZ;
          int h  = hw / WID, w = hw - h * WID;
          int c  = k / KK,  kidx = k - c * KK;
          int dy = kidx / ks - pad;
          int dx = kidx - (kidx / ks) * ks - pad;
          int hy = h + dy, wx = w + dx;
          if (hy >= 0 && hy < HGT && wx >= 0 && wx < WID)
            v = x[((b * Cin + c) * HGT + hy) * WID + wx];
        }
        As[kk * 132 + mi] = v;
      }
    }
    // B tile: 64(K) x 32(N), scalar stage (8 per thread).
    #pragma unroll
    for (int it = 0; it < 8; ++it) {
      int i  = tid + it * 256;
      int kk = i >> 5, ni = i & 31;
      int k = kb + kk, n = ntile + ni;
      Bs[kk * 36 + ni] = (k < Ktot && n < N) ? wgt[n * Ktot + k] : 0.0f;
    }
    if (useAsync) wait_async0();
    __syncthreads();

    // Fragment element e holds K = e + 8*(e>=8) + 8*(lane>=16)  (ISA 7.12.2).
    v16h af, bf0, bf1;
    #pragma unroll
    for (int e = 0; e < 16; ++e) {
      int kk = e + ((e >= 8) ? 8 : 0) + kofs;
      af[e]  = (_Float16)(As[kk * 132 + mrow] * psc);
      bf0[e] = (_Float16)Bs[kk * 36 + nc0];
      bf1[e] = (_Float16)Bs[kk * 36 + nc1];
    }
    acc0 = __builtin_amdgcn_wmma_f32_16x16x32_f16(false, af, false, bf0,
                                                  (short)0, acc0, false, false);
    acc1 = __builtin_amdgcn_wmma_f32_16x16x32_f16(false, af, false, bf1,
                                                  (short)0, acc1, false, false);
    #pragma unroll
    for (int e = 0; e < 16; ++e) {
      int kk = 32 + e + ((e >= 8) ? 8 : 0) + kofs;
      af[e]  = (_Float16)(As[kk * 132 + mrow] * psc);
      bf0[e] = (_Float16)Bs[kk * 36 + nc0];
      bf1[e] = (_Float16)Bs[kk * 36 + nc1];
    }
    acc0 = __builtin_amdgcn_wmma_f32_16x16x32_f16(false, af, false, bf0,
                                                  (short)0, acc0, false, false);
    acc1 = __builtin_amdgcn_wmma_f32_16x16x32_f16(false, af, false, bf1,
                                                  (short)0, acc1, false, false);
    __syncthreads();
  }

  // Epilogue. D layout: VGPR r, lane L -> M = r + 8*(L>=16), N = L%16.
  #pragma unroll
  for (int r = 0; r < 8; ++r) {
    int mi = wv * 16 + r + ((lane >= 16) ? 8 : 0);
    int m = mtile + mi;
    int b = m / HWSZ, hw = m - b * HWSZ;
    #pragma unroll
    for (int half = 0; half < 2; ++half) {
      int n = ntile + (half ? nc1 : nc0);
      if (m < BHW && n < N) {
        float v = half ? acc1[r] : acc0[r];
        if (act == 0)      v = v + (bias ? bias[n] : 0.0f);
        else if (act == 1) v = fmaxf(0.0f, v * gscale[n] + bias[n]);
        else               v = 1.0f / (1.0f + __expf(-(v + (bias ? bias[n] : 0.0f))));
        out[(b * N + n) * HWSZ + hw] = v;
      }
    }
  }
}

// ---------------------------------------------------------------------------
// Deformable-conv sampling einsum as WMMA GEMM.
//   out[b,o,hw] = relu(g[o] * sum_{c,k} sample(x; om)[b,c,k,hw] * wgt[o, c*9+k] + bias[o])
// Per block: precompute bilinear corner indices + mask-scaled weights for the
// 128-pixel x 9-tap tile into LDS, then stage the sampled A tile (64 K x 128)
// as f16 and run four WMMAs per K chunk (two N slices). wgt is [O x C*9].
// ---------------------------------------------------------------------------
__global__ __launch_bounds__(256)
void dcn_wmma_kernel(const float* __restrict__ x, const float* __restrict__ om,
                     const float* __restrict__ wgt, const float* __restrict__ g,
                     const float* __restrict__ bias, float* __restrict__ out,
                     int C, int O)
{
  const int Ktot = C * 9;
  const int mtile = blockIdx.x * 128;

  __shared__ int      pI[1152 * 4];             // 128 pixels x 9 taps x 4 corners
  __shared__ float    pW[1152 * 4];
  __shared__ __align__(16) _Float16 As[64 * 132];   // [kk][mi]
  __shared__ __align__(16) _Float16 Bs[64 * 36];    // [kk][ni]

  const int tid  = threadIdx.x;
  const int lane = tid & 31;
  const int wv   = tid >> 5;
  const int mrow = wv * 16 + (lane & 15);
  const int nc0  = lane & 15;
  const int nc1  = 16 + nc0;
  const int kofs = (lane >= 16) ? 8 : 0;

  // ---- precompute sampling geometry for this pixel tile ----
  for (int i = tid; i < 1152; i += 256) {
    int mi = i / 9, k = i - mi * 9;
    int m = mtile + mi;
    int b = m / HWSZ, hw = m - b * HWSZ;
    int h = hw / WID, w = hw - h * WID;
    const float* omp = om + b * 27 * HWSZ + hw;
    float dy = omp[k * HWSZ];
    float dx = omp[(9 + k) * HWSZ];
    float mk = 1.0f / (1.0f + __expf(-omp[(18 + k) * HWSZ]));
    float py = (float)h + (float)(k / 3 - 1) + dy;
    float px = (float)w + (float)(k % 3 - 1) + dx;
    float y0 = floorf(py), x0 = floorf(px);
    float wy = py - y0, wx = px - x0;
    int iy0 = (int)y0, ix0 = (int)x0;
    int yc0 = min(max(iy0, 0), HGT - 1), yc1 = min(max(iy0 + 1, 0), HGT - 1);
    int xc0 = min(max(ix0, 0), WID - 1), xc1 = min(max(ix0 + 1, 0), WID - 1);
    float vy0 = (iy0 >= 0 && iy0 < HGT) ? 1.f : 0.f;
    float vy1 = (iy0 + 1 >= 0 && iy0 + 1 < HGT) ? 1.f : 0.f;
    float vx0 = (ix0 >= 0 && ix0 < WID) ? 1.f : 0.f;
    float vx1 = (ix0 + 1 >= 0 && ix0 + 1 < WID) ? 1.f : 0.f;
    pI[i * 4 + 0] = yc0 * WID + xc0;  pW[i * 4 + 0] = mk * vy0 * vx0 * (1 - wy) * (1 - wx);
    pI[i * 4 + 1] = yc0 * WID + xc1;  pW[i * 4 + 1] = mk * vy0 * vx1 * (1 - wy) * wx;
    pI[i * 4 + 2] = yc1 * WID + xc0;  pW[i * 4 + 2] = mk * vy1 * vx0 * wy * (1 - wx);
    pI[i * 4 + 3] = yc1 * WID + xc1;  pW[i * 4 + 3] = mk * vy1 * vx1 * wy * wx;
  }
  __syncthreads();

  v8f acc0 = {}, acc1 = {};

  for (int kb = 0; kb < Ktot; kb += 64) {
    // Stage sampled A tile (gathered): 64 K x 128 pixels.
    for (int it = 0; it < 32; ++it) {
      int i  = tid + it * 256;
      int kk = i >> 7, mi = i & 127;
      int kidx = kb + kk;
      _Float16 v = (_Float16)0.0f;
      if (kidx < Ktot) {
        int c = kidx / 9, k9 = kidx - c * 9;
        int m = mtile + mi;
        int b = m / HWSZ;
        const float* xc = x + ((size_t)b * C + c) * HWSZ;
        int e = (mi * 9 + k9) * 4;
        v = (_Float16)(pW[e + 0] * xc[pI[e + 0]] + pW[e + 1] * xc[pI[e + 1]] +
                       pW[e + 2] * xc[pI[e + 2]] + pW[e + 3] * xc[pI[e + 3]]);
      }
      As[kk * 132 + mi] = v;
    }
    // Stage B tile: 64 K x 32 N (N = output channels O, zero-padded).
    #pragma unroll
    for (int it = 0; it < 8; ++it) {
      int i  = tid + it * 256;
      int kk = i >> 5, ni = i & 31;
      int kidx = kb + kk;
      Bs[kk * 36 + ni] = (kidx < Ktot && ni < O)
                         ? (_Float16)wgt[ni * Ktot + kidx] : (_Float16)0.0f;
    }
    __syncthreads();

    v16h af, bf0, bf1;
    #pragma unroll
    for (int e = 0; e < 16; ++e) {
      int kk = e + ((e >= 8) ? 8 : 0) + kofs;
      af[e]  = As[kk * 132 + mrow];
      bf0[e] = Bs[kk * 36 + nc0];
      bf1[e] = Bs[kk * 36 + nc1];
    }
    acc0 = __builtin_amdgcn_wmma_f32_16x16x32_f16(false, af, false, bf0,
                                                  (short)0, acc0, false, false);
    acc1 = __builtin_amdgcn_wmma_f32_16x16x32_f16(false, af, false, bf1,
                                                  (short)0, acc1, false, false);
    #pragma unroll
    for (int e = 0; e < 16; ++e) {
      int kk = 32 + e + ((e >= 8) ? 8 : 0) + kofs;
      af[e]  = As[kk * 132 + mrow];
      bf0[e] = Bs[kk * 36 + nc0];
      bf1[e] = Bs[kk * 36 + nc1];
    }
    acc0 = __builtin_amdgcn_wmma_f32_16x16x32_f16(false, af, false, bf0,
                                                  (short)0, acc0, false, false);
    acc1 = __builtin_amdgcn_wmma_f32_16x16x32_f16(false, af, false, bf1,
                                                  (short)0, acc1, false, false);
    __syncthreads();
  }

  #pragma unroll
  for (int r = 0; r < 8; ++r) {
    int mi = wv * 16 + r + ((lane >= 16) ? 8 : 0);
    int m = mtile + mi;
    int b = m / HWSZ, hw = m - b * HWSZ;
    if (nc0 < O)
      out[(b * O + nc0) * HWSZ + hw] = fmaxf(0.0f, acc0[r] * g[nc0] + bias[nc0]);
    if (nc1 < O)
      out[(b * O + nc1) * HWSZ + hw] = fmaxf(0.0f, acc1[r] * g[nc1] + bias[nc1]);
  }
}

// ---------- small elementwise kernels ----------
__global__ void copy_ch_kernel(const float* __restrict__ src, float* __restrict__ dst,
                               int srcC, int dstC, int coff, int total)
{
  int i = blockIdx.x * blockDim.x + threadIdx.x;
  if (i >= total) return;
  int b = i / (srcC * HWSZ);
  int r = i - b * srcC * HWSZ;
  int c = r / HWSZ;
  int hw = r - c * HWSZ;
  dst[(b * dstC + coff + c) * HWSZ + hw] = src[i];
}

__global__ void gate1_kernel(const float* __restrict__ x, const float* __restrict__ w,
                             const float* __restrict__ b, float* __restrict__ out, int C)
{
  int m = blockIdx.x * blockDim.x + threadIdx.x;
  if (m >= BHW) return;
  int bb = m / HWSZ, hw = m - bb * HWSZ;
  const float* xp = x + bb * C * HWSZ + hw;
  float s = b[0];
  for (int c = 0; c < C; ++c) s += xp[c * HWSZ] * w[c];
  out[m] = 1.0f / (1.0f + __expf(-s));
}

__global__ void pdep_comb_kernel(const float* __restrict__ y, const float* __restrict__ A,
                                 const float* __restrict__ Bt, float* __restrict__ out, int n)
{
  int i = blockIdx.x * blockDim.x + threadIdx.x;
  if (i >= n) return;
  int b = i / (HID * HWSZ);
  int hw = i % HWSZ;
  int m = b * HWSZ + hw;
  out[i] = (1.0f - A[m]) * y[i] * Bt[m];
}

__global__ void gru_gate_kernel(const float* __restrict__ ih, const float* __restrict__ bhh,
                                float* __restrict__ out, int n)
{
  int i = blockIdx.x * blockDim.x + threadIdx.x;
  if (i >= n) return;
  int b = i / (HID * HWSZ);
  int r = i - b * HID * HWSZ;
  int j = r / HWSZ;
  int hw = r - j * HWSZ;
  const float* ihp = ih + b * 3 * HID * HWSZ + hw;
  float rg = 1.0f / (1.0f + __expf(-(ihp[j * HWSZ] + bhh[j])));
  float zg = 1.0f / (1.0f + __expf(-(ihp[(HID + j) * HWSZ] + bhh[HID + j])));
  float ng = tanhf(ihp[(2 * HID + j) * HWSZ] + rg * bhh[2 * HID + j]);
  out[i] = (1.0f - zg) * ng;
}

__global__ void add3_kernel(const float* a, const float* b, const float* c, float* o, int n)
{ int i = blockIdx.x * blockDim.x + threadIdx.x; if (i < n) o[i] = a[i] + b[i] + c[i]; }
__global__ void add2_kernel(const float* a, const float* b, float* o, int n)
{ int i = blockIdx.x * blockDim.x + threadIdx.x; if (i < n) o[i] = a[i] + b[i]; }
__global__ void emax_kernel(float* acc, const float* v, int n)
{ int i = blockIdx.x * blockDim.x + threadIdx.x; if (i < n) acc[i] = fmaxf(acc[i], v[i]); }

// ===========================================================================
// Host side
// ===========================================================================
struct Arena {
  char* base; size_t off;
  float* alloc(size_t n) {
    float* p = (float*)(base + off);
    off = (off + n * sizeof(float) + 255) & ~(size_t)255;
    return p;
  }
};

struct CBR  { const float *b, *g, *w; };
struct Seq  { CBR c1, c2; };
struct Dec  { const float* b2; CBR c1, c3; const float* w2; };
struct Dcn  { const float *b, *g, *ob, *ow, *w; };
struct Pdep { const float *Ab, *Aw, *Bb, *Bw; Dcn d1, d2; };
struct GruL { const float *bhh, *bih, *whh, *wih; };
struct Gru  { GruL l0, l1; };

// Parameter leaves arrive flattened in jax tree order (dict keys sorted).
struct Cur {
  void* const* a; int i;
  const float* n() { return (const float*)a[i++]; }
};
static CBR  ldCBR (Cur& c) { CBR  r; r.b = c.n(); r.g = c.n(); r.w = c.n(); return r; }
static Seq  ldSeq (Cur& c) { Seq  r; r.c1 = ldCBR(c); r.c2 = ldCBR(c); return r; }
static Dec  ldDec (Cur& c) { Dec  r; r.b2 = c.n(); r.c1 = ldCBR(c); r.c3 = ldCBR(c); r.w2 = c.n(); return r; }
static Dcn  ldDcn (Cur& c) { Dcn  r; r.b = c.n(); r.g = c.n(); r.ob = c.n(); r.ow = c.n(); r.w = c.n(); return r; }
static Pdep ldPdep(Cur& c) { Pdep r; r.Ab = c.n(); r.Aw = c.n(); r.Bb = c.n(); r.Bw = c.n(); r.d1 = ldDcn(c); r.d2 = ldDcn(c); return r; }
static GruL ldGruL(Cur& c) { GruL r; r.bhh = c.n(); r.bih = c.n(); r.whh = c.n(); r.wih = c.n(); return r; }
static Gru  ldGru (Cur& c) { Gru  r; r.l0 = ldGruL(c); r.l1 = ldGruL(c); return r; }

static void run_conv(hipStream_t s, const float* x, const float* w, const float* g,
                     const float* b, const float* pre, float* out,
                     int Cin, int N, int ks, int act)
{
  dim3 grid(ceil_div(BHW, 128), ceil_div(N, 32));
  conv_wmma_kernel<<<grid, 256, 0, s>>>(x, w, g, b, pre, out, Cin, N, ks, act);
}
static void run_cbr(hipStream_t s, const float* x, const CBR& p, int Cin, int N, int ks,
                    float* out, const float* pre = nullptr)
{ run_conv(s, x, p.w, p.g, p.b, pre, out, Cin, N, ks, 1); }

static void cat_ch(hipStream_t s, float* dst, int dstC, int coff, const float* src, int srcC)
{
  int tot = BATCH * srcC * HWSZ;
  copy_ch_kernel<<<ceil_div(tot, 256), 256, 0, s>>>(src, dst, srcC, dstC, coff, tot);
}

static void run_seq(hipStream_t s, const float* x, int Cin, const Seq& p, int ks,
                    float* out, Arena ar)
{
  float* mid = ar.alloc((size_t)BATCH * 2 * HID * HWSZ);
  run_cbr(s, x, p.c1, Cin, 2 * HID, ks, mid);
  run_cbr(s, mid, p.c2, 2 * HID, HID, ks, out);
}

static void run_gru2(hipStream_t s, const float* in, int Cin, const Gru& g, float* out, Arena ar)
{
  float* ih = ar.alloc((size_t)BATCH * 3 * HID * HWSZ);
  float* t  = ar.alloc((size_t)BATCH * HID * HWSZ);
  const int n = BATCH * HID * HWSZ;
  run_conv(s, in, g.l0.wih, nullptr, g.l0.bih, nullptr, ih, Cin, 3 * HID, 1, 0);
  gru_gate_kernel<<<ceil_div(n, 256), 256, 0, s>>>(ih, g.l0.bhh, t, n);
  run_conv(s, t, g.l1.wih, nullptr, g.l1.bih, nullptr, ih, HID, 3 * HID, 1, 0);
  gru_gate_kernel<<<ceil_div(n, 256), 256, 0, s>>>(ih, g.l1.bhh, out, n);
}

static void run_dcn(hipStream_t s, const float* x, int Cin, int O, const Dcn& p,
                    float* out, Arena ar)
{
  float* om = ar.alloc((size_t)BATCH * 27 * HWSZ);
  run_conv(s, x, p.ow, nullptr, p.ob, nullptr, om, Cin, 27, 3, 0);
  dim3 grid(ceil_div(BHW, 128), 1);
  dcn_wmma_kernel<<<grid, 256, 0, s>>>(x, om, p.w, p.g, p.b, out, Cin, O);
}

static void run_pdep(hipStream_t s, const Pdep& p, const float* fea,
                     const float* pA, const float* pB, float* out, Arena ar)
{
  float* cat = ar.alloc((size_t)BATCH * (IND + HID) * HWSZ);
  cat_ch(s, cat, IND + HID, 0, fea, IND);
  cat_ch(s, cat, IND + HID, IND, pA, HID);
  float* y1 = ar.alloc((size_t)BATCH * 2 * HID * HWSZ);
  run_dcn(s, cat, IND + HID, 2 * HID, p.d1, y1, ar);
  float* y2 = ar.alloc((size_t)BATCH * HID * HWSZ);
  run_dcn(s, y1, 2 * HID, HID, p.d2, y2, ar);
  float* A  = ar.alloc(BHW);
  float* Bt = ar.alloc(BHW);
  gate1_kernel<<<ceil_div(BHW, 256), 256, 0, s>>>(pA, p.Aw, p.Ab, A, HID);
  gate1_kernel<<<ceil_div(BHW, 256), 256, 0, s>>>(pB, p.Bw, p.Bb, Bt, HID);
  const int n = BATCH * HID * HWSZ;
  pdep_comb_kernel<<<ceil_div(n, 256), 256, 0, s>>>(y2, A, Bt, out, n);
}

static void run_decomp(hipStream_t s, const Dec& p, const float* fea,
                       const float* a, const float* b, float* out, Arena ar)
{
  float* cat = ar.alloc((size_t)BATCH * 2 * HID * HWSZ);
  cat_ch(s, cat, 2 * HID, 0, a, HID);
  cat_ch(s, cat, 2 * HID, HID, b, HID);
  float* t = ar.alloc((size_t)BATCH * 2 * HID * HWSZ);
  run_cbr(s, cat, p.c1, 2 * HID, 2 * HID, 1, t);
  float* att = ar.alloc(BHW);
  gate1_kernel<<<ceil_div(BHW, 256), 256, 0, s>>>(t, p.w2, p.b2, att, 2 * HID);
  run_cbr(s, fea, p.c3, IND, HID, 1, out, att);   // att*fea fused via prescale (async A path)
}

extern "C" void kernel_launch(void* const* d_in, const int* in_sizes, int n_in,
                              void* d_out, int out_size, void* d_ws, size_t ws_size,
                              hipStream_t stream)
{
  (void)in_sizes; (void)n_in; (void)out_size; (void)ws_size;
  hipStream_t s = stream;
  const float* p_fea = (const float*)d_in[0];
  const float* h_fea = (const float*)d_in[1];
  const float* xf    = (const float*)d_in[2];
  const float* xh    = (const float*)d_in[3];
  const float* xp    = (const float*)d_in[4];

  Cur cur{d_in, 5};
  // params: 'full' < 'half' < 'part'
  Seq  f_conv_hf = ldSeq(cur);  Gru f_gru = ldGru(cur);
  Seq  h_conv_phl = ldSeq(cur), h_conv_phu = ldSeq(cur);
  Pdep h_dp_h = ldPdep(cur),    h_dp_l = ldPdep(cur);
  Dec  h_fhl = ldDec(cur),      h_fhu = ldDec(cur);
  Gru  h_gru_l = ldGru(cur),    h_gru_u = ldGru(cur);
  Gru  p_gru[6];  for (int i = 0; i < 6; ++i)  p_gru[i] = ldGru(cur);
  Dec  p_hp[6];   for (int i = 0; i < 6; ++i)  p_hp[i]  = ldDec(cur);
  Pdep p_pdp[10]; for (int i = 0; i < 10; ++i) p_pdp[i] = ldPdep(cur);

  const size_t CH1 = (size_t)BATCH * HID * HWSZ;
  const float* xh0 = xh;
  const float* xh1 = xh + CH1;
  const float* xps[6];
  for (int j = 0; j < 6; ++j) xps[j] = xp + (size_t)j * CH1;

  float* out_xp = (float*)d_out;            // [6,B,HID,H,W]
  float* out_xh = out_xp + 6 * CH1;         // [2,B,HID,H,W]
  float* out_xf = out_xh + 2 * CH1;         // [B,HID,H,W]

  Arena ar{(char*)d_ws, 0};
  const int n10 = (int)CH1;

  // ---- full branch ----
  {
    Arena a = ar;
    float* cat30 = a.alloc((size_t)BATCH * 3 * HID * HWSZ);
    cat_ch(s, cat30, 3 * HID, 0, xh0, HID);
    cat_ch(s, cat30, 3 * HID, HID, xh1, HID);
    cat_ch(s, cat30, 3 * HID, 2 * HID, xf, HID);
    float* fmsg = a.alloc(CH1);
    run_seq(s, cat30, 3 * HID, f_conv_hf, 3, fmsg, a);
    float* cat20 = a.alloc(2 * CH1);
    cat_ch(s, cat20, 2 * HID, 0, xf, HID);
    cat_ch(s, cat20, 2 * HID, HID, fmsg, HID);
    run_gru2(s, cat20, 2 * HID, f_gru, out_xf, a);
  }

  // ---- half branch: upper ----
  {
    Arena a = ar;
    float* cat50 = a.alloc((size_t)BATCH * 5 * HID * HWSZ);
    for (int j = 0; j < 4; ++j) cat_ch(s, cat50, 5 * HID, j * HID, xps[j], HID);
    cat_ch(s, cat50, 5 * HID, 4 * HID, xh0, HID);
    float* xphu = a.alloc(CH1);
    run_seq(s, cat50, 5 * HID, h_conv_phu, 3, xphu, a);
    float* xlh = a.alloc(CH1);
    run_pdep(s, h_dp_h, h_fea, xh1, xh0, xlh, a);
    float* xfhu = a.alloc(CH1);
    run_decomp(s, h_fhu, h_fea, xf, xh0, xfhu, a);
    float* sum = a.alloc(CH1);
    add3_kernel<<<ceil_div(n10, 256), 256, 0, s>>>(xphu, xlh, xfhu, sum, n10);
    float* cat20 = a.alloc(2 * CH1);
    cat_ch(s, cat20, 2 * HID, 0, xh0, HID);
    cat_ch(s, cat20, 2 * HID, HID, sum, HID);
    run_gru2(s, cat20, 2 * HID, h_gru_u, out_xh, a);
  }

  // ---- half branch: lower ----
  {
    Arena a = ar;
    float* cat30 = a.alloc((size_t)BATCH * 3 * HID * HWSZ);
    cat_ch(s, cat30, 3 * HID, 0, xps[4], HID);
    cat_ch(s, cat30, 3 * HID, HID, xps[5], HID);
    cat_ch(s, cat30, 3 * HID, 2 * HID, xh1, HID);
    float* xphl = a.alloc(CH1);
    run_seq(s, cat30, 3 * HID, h_conv_phl, 3, xphl, a);
    float* xuh = a.alloc(CH1);
    run_pdep(s, h_dp_l, h_fea, xh0, xh1, xuh, a);
    float* xfhl = a.alloc(CH1);
    run_decomp(s, h_fhl, h_fea, xf, xh1, xfhl, a);
    float* sum = a.alloc(CH1);
    add3_kernel<<<ceil_div(n10, 256), 256, 0, s>>>(xphl, xuh, xfhl, sum, n10);
    float* cat20 = a.alloc(2 * CH1);
    cat_ch(s, cat20, 2 * HID, 0, xh1, HID);
    cat_ch(s, cat20, 2 * HID, HID, sum, HID);
    run_gru2(s, cat20, 2 * HID, h_gru_l, out_xh + CH1, a);
  }

  // ---- part branch ----
  {
    static const int EA[10] = {0, 1, 1, 1, 2, 2, 3, 4, 4, 5};
    static const int EB[10] = {1, 0, 2, 4, 1, 3, 2, 1, 5, 4};
    Arena a = ar;
    float* buckets[6];
    bool   has[6] = {false, false, false, false, false, false};
    for (int j = 0; j < 6; ++j) buckets[j] = a.alloc(CH1);
    float* tmp = a.alloc(CH1);
    for (int e = 0; e < 10; ++e) {
      Arena a2 = a;
      run_pdep(s, p_pdp[e], p_fea, xps[EA[e]], xps[EB[e]], tmp, a2);
      int bkt = EB[e];
      if (!has[bkt]) {
        hipMemcpyAsync(buckets[bkt], tmp, CH1 * sizeof(float),
                       hipMemcpyDeviceToDevice, s);
        has[bkt] = true;
      } else {
        emax_kernel<<<ceil_div(n10, 256), 256, 0, s>>>(buckets[bkt], tmp, n10);
      }
    }
    for (int j = 0; j < 6; ++j) {
      Arena a2 = a;
      const float* src = (j <= 3) ? xh0 : xh1;   // UPPER = {1,2,3,4} -> indices 0..3
      float* xhp = a2.alloc(CH1);
      run_decomp(s, p_hp[j], p_fea, src, xps[j], xhp, a2);
      float* sum = a2.alloc(CH1);
      add2_kernel<<<ceil_div(n10, 256), 256, 0, s>>>(buckets[j], xhp, sum, n10);
      float* cat20 = a2.alloc(2 * CH1);
      cat_ch(s, cat20, 2 * HID, 0, xps[j], HID);
      cat_ch(s, cat20, 2 * HID, HID, sum, HID);
      run_gru2(s, cat20, 2 * HID, p_gru[j], out_xp + (size_t)j * CH1, a2);
    }
  }
}
